// WeightOnlyQuantizedLinear_89318139887971
// MI455X (gfx1250) — matrix operand order repaired and checked by reading, not compiled, not run on hardware
//
#include <hip/hip_runtime.h>

typedef __attribute__((ext_vector_type(16))) _Float16 v16h;
typedef __attribute__((ext_vector_type(8)))  _Float16 v8h;
typedef __attribute__((ext_vector_type(8)))  float    v8f;
typedef __attribute__((ext_vector_type(4)))  int      i32x4;

#define BM 128
#define BN 128
#define BK 32
#define PADH 8
#define LDSTR  (BK + PADH)    // fused path: 40 halves/row (80B, 16B-multiple)
#define BK2 64
#define LDSTR2 (BK2 + PADH)   // f16 path: 72 halves/row (144B, 16B-multiple)

union F16x16 { v16h v; v8h h[2]; };

// ---- optional CDNA5 async global->LDS staging --------------------------------
#if defined(__has_builtin)
#if __has_builtin(__builtin_amdgcn_global_load_async_to_lds_b128) && \
    __has_builtin(__builtin_amdgcn_s_wait_asynccnt)
#define HAVE_ASYNC_LDS 1
#endif
#endif
#ifndef HAVE_ASYNC_LDS
#define HAVE_ASYNC_LDS 0
#endif

#if HAVE_ASYNC_LDS
// builtin signature (from clang diagnostic): (int4 AS1* src, int4 AS3* dst, imm off, imm cpol)
// generic->AS1 is numerically identity; generic LDS ptr keeps offset in low 32b
#define GPTR(p) ((__attribute__((address_space(1))) i32x4*)(unsigned long long)(p))
#define LPTR(p) ((__attribute__((address_space(3))) i32x4*)(unsigned int)(unsigned long long)(p))

__device__ __forceinline__ void stage_async_tile(const _Float16* xp, const _Float16* wp,
                                                 _Float16* da, _Float16* db) {
#pragma unroll
    for (int u = 0; u < 4; ++u) {
        __builtin_amdgcn_global_load_async_to_lds_b128(GPTR(xp + 8 * u), LPTR(da + 8 * u), 0, 0);
        __builtin_amdgcn_global_load_async_to_lds_b128(GPTR(wp + 8 * u), LPTR(db + 8 * u), 0, 0);
    }
}
#endif

// ---------------------------------------------------------------------------
// Pass 1a: dequantize W (int32 uint8-range) -> f16, one row per block
// ---------------------------------------------------------------------------
__global__ __launch_bounds__(256)
void dequant_w_kernel(const int* __restrict__ qw,
                      const float* __restrict__ scales,
                      const int* __restrict__ zps,
                      _Float16* __restrict__ wh, int K)
{
    const int n = blockIdx.x;
    const float scl = scales[n];
    const float zpf = (float)zps[n];
    const int* src = qw + (size_t)n * K;
    _Float16*  dst = wh + (size_t)n * K;
    for (int k0 = threadIdx.x * 16; k0 < K; k0 += blockDim.x * 16) {
        int4 q[4];
#pragma unroll
        for (int u = 0; u < 4; ++u) q[u] = ((const int4*)(src + k0))[u];
        alignas(16) _Float16 h[16];
#pragma unroll
        for (int u = 0; u < 4; ++u) {
            h[4*u+0] = (_Float16)(((float)q[u].x - zpf) * scl);
            h[4*u+1] = (_Float16)(((float)q[u].y - zpf) * scl);
            h[4*u+2] = (_Float16)(((float)q[u].z - zpf) * scl);
            h[4*u+3] = (_Float16)(((float)q[u].w - zpf) * scl);
        }
        v8h* d = (v8h*)(dst + k0);
        d[0] = *(const v8h*)&h[0];
        d[1] = *(const v8h*)&h[8];
    }
}

// ---------------------------------------------------------------------------
// Pass 1b: convert X f32 -> f16, one row per block
// ---------------------------------------------------------------------------
__global__ __launch_bounds__(256)
void convert_x_kernel(const float* __restrict__ x,
                      _Float16* __restrict__ xh, int K)
{
    const int m = blockIdx.x;
    const float* src = x  + (size_t)m * K;
    _Float16*    dst = xh + (size_t)m * K;
    for (int k0 = threadIdx.x * 16; k0 < K; k0 += blockDim.x * 16) {
        float4 f[4];
#pragma unroll
        for (int u = 0; u < 4; ++u) f[u] = ((const float4*)(src + k0))[u];
        alignas(16) _Float16 h[16];
#pragma unroll
        for (int u = 0; u < 4; ++u) {
            h[4*u+0] = (_Float16)f[u].x;
            h[4*u+1] = (_Float16)f[u].y;
            h[4*u+2] = (_Float16)f[u].z;
            h[4*u+3] = (_Float16)f[u].w;
        }
        v8h* d = (v8h*)(dst + k0);
        d[0] = *(const v8h*)&h[0];
        d[1] = *(const v8h*)&h[8];
    }
}

// ---------------------------------------------------------------------------
// Pass 2: pure f16 GEMM with BK=64, 16 WMMAs per wave per barrier.
// Staging: async global->LDS DMA when available, else regs + ds_store.
// ---------------------------------------------------------------------------
__global__ __launch_bounds__(256)
void gemm_f16_wmma(const _Float16* __restrict__ xh,
                   const _Float16* __restrict__ wh,
                   const float* __restrict__ bias,
                   float* __restrict__ out,
                   int M, int N, int K)
{
    __shared__ _Float16 lA[2][BM * LDSTR2];
    __shared__ _Float16 lB[2][BN * LDSTR2];

    const int tid  = threadIdx.x;
    const int lane = tid & 31;
    const int wave = tid >> 5;
    const int wm   = wave & 1;
    const int wn   = wave >> 1;

    const int bm = blockIdx.y * BM;
    const int bn = blockIdx.x * BN;

    // staging: 2 threads per row, 32 contiguous halves (64B) each
    const int row  = tid >> 1;
    const int koff = (tid & 1) * 32;

    const _Float16* xrow = xh + (size_t)(bm + row) * K + koff;
    const _Float16* wrow = wh + (size_t)(bn + row) * K + koff;

    _Float16* ldsA0 = &lA[0][row * LDSTR2 + koff];
    _Float16* ldsB0 = &lB[0][row * LDSTR2 + koff];
    _Float16* ldsA1 = &lA[1][row * LDSTR2 + koff];
    _Float16* ldsB1 = &lB[1][row * LDSTR2 + koff];

    v8f acc[4][2];
#pragma unroll
    for (int i = 0; i < 4; ++i)
#pragma unroll
        for (int j = 0; j < 2; ++j)
#pragma unroll
            for (int r = 0; r < 8; ++r) acc[i][j][r] = 0.0f;

    const int KT = K / BK2;

#if HAVE_ASYNC_LDS
    stage_async_tile(xrow, wrow, ldsA0, ldsB0);
    __builtin_amdgcn_s_wait_asynccnt(0);
    __syncthreads();
#else
    v8h xa[4], wb[4];
#pragma unroll
    for (int u = 0; u < 4; ++u) {
        xa[u] = ((const v8h*)xrow)[u];
        wb[u] = ((const v8h*)wrow)[u];
    }
#pragma unroll
    for (int u = 0; u < 4; ++u) {
        ((v8h*)ldsA0)[u] = xa[u];
        ((v8h*)ldsB0)[u] = wb[u];
    }
    __syncthreads();
#endif

    const int a_m  = (lane & 15);
    const int a_kb = (lane >> 4) * 8;    // A: K 0-7/16-23 vs 8-15/24-31 by lane half
    const int b_kb = (lane >> 4) * 8;    // B: K 0-7/16-23 vs 8-15/24-31 by lane half

    for (int kt = 0; kt < KT; ++kt) {
        const int cur = kt & 1;
        const bool more = (kt + 1) < KT;

#if HAVE_ASYNC_LDS
        if (more) {
            const _Float16* xp = xrow + (size_t)(kt + 1) * BK2;
            const _Float16* wp = wrow + (size_t)(kt + 1) * BK2;
            __builtin_prefetch(xp + BK2, 0, 1);   // global_prefetch for kt+2
            __builtin_prefetch(wp + BK2, 0, 1);
            stage_async_tile(xp, wp, cur ? ldsA0 : ldsA1, cur ? ldsB0 : ldsB1);
        }
#else
        if (more) {
            const _Float16* xp = xrow + (size_t)(kt + 1) * BK2;
            const _Float16* wp = wrow + (size_t)(kt + 1) * BK2;
            __builtin_prefetch(xp + BK2, 0, 1);
            __builtin_prefetch(wp + BK2, 0, 1);
#pragma unroll
            for (int u = 0; u < 4; ++u) {
                xa[u] = ((const v8h*)xp)[u];
                wb[u] = ((const v8h*)wp)[u];
            }
        }
#endif

#pragma unroll
        for (int kc = 0; kc < 2; ++kc) {
            const int kb = kc * 32;
            F16x16 afr[4];
#pragma unroll
            for (int i = 0; i < 4; ++i) {
                const int ar = wm * 64 + i * 16 + a_m;
                const _Float16* base = &lA[cur][ar * LDSTR2 + kb];
                afr[i].h[0] = *(const v8h*)&base[a_kb];
                afr[i].h[1] = *(const v8h*)&base[16 + a_kb];
            }
            F16x16 bfr[2];
#pragma unroll
            for (int j = 0; j < 2; ++j) {
                const int br = wn * 32 + j * 16 + a_m;
                const _Float16* base = &lB[cur][br * LDSTR2 + kb];
                bfr[j].h[0] = *(const v8h*)&base[b_kb];
                bfr[j].h[1] = *(const v8h*)&base[16 + b_kb];
            }
#pragma unroll
            for (int i = 0; i < 4; ++i)
#pragma unroll
                for (int j = 0; j < 2; ++j)
                    acc[i][j] = __builtin_amdgcn_wmma_f32_16x16x32_f16(
                        false, afr[i].v, false, bfr[j].v,
                        (short)0, acc[i][j], false, false);
        }

#if HAVE_ASYNC_LDS
        if (more) __builtin_amdgcn_s_wait_asynccnt(0);
        __syncthreads();
#else
        if (more) {
            _Float16* da = cur ? ldsA0 : ldsA1;
            _Float16* db = cur ? ldsB0 : ldsB1;
#pragma unroll
            for (int u = 0; u < 4; ++u) {
                ((v8h*)da)[u] = xa[u];
                ((v8h*)db)[u] = wb[u];
            }
        }
        __syncthreads();
#endif
    }

#pragma unroll
    for (int j = 0; j < 2; ++j) {
        const int n  = bn + wn * 32 + j * 16 + (lane & 15);
        const float bv = bias[n];
#pragma unroll
        for (int i = 0; i < 4; ++i) {
            const int m0 = bm + wm * 64 + i * 16 + ((lane >> 4) << 3);
#pragma unroll
            for (int r = 0; r < 8; ++r)
                out[(size_t)(m0 + r) * N + n] = acc[i][j][r] + bv;
        }
    }
}

// ---------------------------------------------------------------------------
// Fallback: fused dequant + GEMM (used when d_ws is too small)
// ---------------------------------------------------------------------------
__device__ __forceinline__ void stage_regs(const float* __restrict__ xp,
                                           const int*   __restrict__ qp,
                                           float4 xr[4], int4 qr[4]) {
#pragma unroll
    for (int u = 0; u < 4; ++u) {
        xr[u] = ((const float4*)xp)[u];
        qr[u] = ((const int4*)qp)[u];
    }
}

__device__ __forceinline__ void stage_lds(_Float16* __restrict__ la,
                                          _Float16* __restrict__ lb,
                                          int row, int koff,
                                          const float4 xr[4], const int4 qr[4],
                                          float zpf, float scl) {
    alignas(16) _Float16 ha[16];
    alignas(16) _Float16 hb[16];
#pragma unroll
    for (int u = 0; u < 4; ++u) {
        ha[4*u+0] = (_Float16)xr[u].x;
        ha[4*u+1] = (_Float16)xr[u].y;
        ha[4*u+2] = (_Float16)xr[u].z;
        ha[4*u+3] = (_Float16)xr[u].w;
        hb[4*u+0] = (_Float16)(((float)qr[u].x - zpf) * scl);
        hb[4*u+1] = (_Float16)(((float)qr[u].y - zpf) * scl);
        hb[4*u+2] = (_Float16)(((float)qr[u].z - zpf) * scl);
        hb[4*u+3] = (_Float16)(((float)qr[u].w - zpf) * scl);
    }
    v8h* da = (v8h*)&la[row * LDSTR + koff];
    da[0] = *(const v8h*)&ha[0];
    da[1] = *(const v8h*)&ha[8];
    v8h* db = (v8h*)&lb[row * LDSTR + koff];
    db[0] = *(const v8h*)&hb[0];
    db[1] = *(const v8h*)&hb[8];
}

__global__ __launch_bounds__(256)
void wq_linear_wmma(const float* __restrict__ x,
                    const int*   __restrict__ qw,
                    const float* __restrict__ scales,
                    const int*   __restrict__ zps,
                    const float* __restrict__ bias,
                    float* __restrict__ out,
                    int M, int N, int K)
{
    __shared__ _Float16 lA[2][BM * LDSTR];
    __shared__ _Float16 lB[2][BN * LDSTR];

    const int tid  = threadIdx.x;
    const int lane = tid & 31;
    const int wave = tid >> 5;
    const int wm   = wave & 1;
    const int wn   = wave >> 1;

    const int bm = blockIdx.y * BM;
    const int bn = blockIdx.x * BN;

    const int row  = tid >> 1;
    const int koff = (tid & 1) * 16;

    const int   nq  = bn + row;
    const float scl = scales[nq];
    const float zpf = (float)zps[nq];

    const float* xrow = x  + (size_t)(bm + row) * K + koff;
    const int*   qrow = qw + (size_t)nq        * K + koff;

    v8f acc[4][2];
#pragma unroll
    for (int i = 0; i < 4; ++i)
#pragma unroll
        for (int j = 0; j < 2; ++j)
#pragma unroll
            for (int r = 0; r < 8; ++r) acc[i][j][r] = 0.0f;

    const int KT = K / BK;

    float4 xr[4];
    int4   qr[4];
    stage_regs(xrow, qrow, xr, qr);
    stage_lds(&lA[0][0], &lB[0][0], row, koff, xr, qr, zpf, scl);
    __syncthreads();

    const int a_m  = (lane & 15);
    const int a_kb = (lane >> 4) * 8;
    const int b_kb = (lane >> 4) * 8;

    for (int kt = 0; kt < KT; ++kt) {
        const int cur = kt & 1;
        const bool more = (kt + 1) < KT;
        if (more) {
            const float* xp = xrow + (size_t)(kt + 1) * BK;
            const int*   qp = qrow + (size_t)(kt + 1) * BK;
            __builtin_prefetch(xp + BK, 0, 1);
            __builtin_prefetch(qp + BK, 0, 1);
            stage_regs(xp, qp, xr, qr);
        }

        F16x16 afr[4];
#pragma unroll
        for (int i = 0; i < 4; ++i) {
            const int ar = wm * 64 + i * 16 + a_m;
            const _Float16* base = &lA[cur][ar * LDSTR];
            afr[i].h[0] = *(const v8h*)&base[a_kb];
            afr[i].h[1] = *(const v8h*)&base[16 + a_kb];
        }
        F16x16 bfr[2];
#pragma unroll
        for (int j = 0; j < 2; ++j) {
            const int br = wn * 32 + j * 16 + a_m;
            const _Float16* base = &lB[cur][br * LDSTR];
            bfr[j].h[0] = *(const v8h*)&base[b_kb];
            bfr[j].h[1] = *(const v8h*)&base[16 + b_kb];
        }
#pragma unroll
        for (int i = 0; i < 4; ++i)
#pragma unroll
            for (int j = 0; j < 2; ++j)
                acc[i][j] = __builtin_amdgcn_wmma_f32_16x16x32_f16(
                    false, afr[i].v, false, bfr[j].v,
                    (short)0, acc[i][j], false, false);

        if (more)
            stage_lds(&lA[cur ^ 1][0], &lB[cur ^ 1][0], row, koff, xr, qr, zpf, scl);
        __syncthreads();
    }

#pragma unroll
    for (int j = 0; j < 2; ++j) {
        const int n  = bn + wn * 32 + j * 16 + (lane & 15);
        const float bv = bias[n];
#pragma unroll
        for (int i = 0; i < 4; ++i) {
            const int m0 = bm + wm * 64 + i * 16 + ((lane >> 4) << 3);
#pragma unroll
            for (int r = 0; r < 8; ++r)
                out[(size_t)(m0 + r) * N + n] = acc[i][j][r] + bv;
        }
    }
}

// ---------------------------------------------------------------------------
extern "C" void kernel_launch(void* const* d_in, const int* in_sizes, int n_in,
                              void* d_out, int out_size, void* d_ws, size_t ws_size,
                              hipStream_t stream) {
    const float* x      = (const float*)d_in[0];
    const int*   qw     = (const int*)  d_in[1];
    const float* scales = (const float*)d_in[2];
    const int*   zps    = (const int*)  d_in[3];
    const float* bias   = (const float*)d_in[4];
    float* out = (float*)d_out;

    const int N = in_sizes[2];            // 4096
    const int K = in_sizes[1] / N;        // 4096
    const int M = in_sizes[0] / K;        // B*S = 4096

    const size_t need = ((size_t)N * K + (size_t)M * K) * sizeof(_Float16);
    dim3 grid(N / BN, M / BM);

    if (ws_size >= need) {
        _Float16* wh = (_Float16*)d_ws;
        _Float16* xh = (_Float16*)((char*)d_ws + (size_t)N * K * sizeof(_Float16));
        dequant_w_kernel<<<N, 256, 0, stream>>>(qw, scales, zps, wh, K);
        convert_x_kernel<<<M, 256, 0, stream>>>(x, xh, K);
        gemm_f16_wmma<<<grid, 256, 0, stream>>>(xh, wh, bias, out, M, N, K);
    } else {
        wq_linear_wmma<<<grid, 256, 0, stream>>>(x, qw, scales, zps, bias, out, M, N, K);
    }
}
